// XEyTransformerLayer_78116865179713
// MI455X (gfx1250) — matrix operand order repaired and checked by reading, not compiled, not run on hardware
//
#include <hip/hip_runtime.h>

// ---------------------------------------------------------------------------
// DiGress XEyTransformerLayer for MI455X (gfx1250), wave32 + WMMA bf16.
// Round 3: transposed Y tile (vector LDS ops in softmax), v_cvt_pk_bf16_f32
// packed conversions, fully vectorized A-staging, per-batch scaled e_out pack.
// ---------------------------------------------------------------------------

typedef __attribute__((ext_vector_type(16))) __bf16 v16bf;
typedef __attribute__((ext_vector_type(8)))  float  v8f;

union FragU { unsigned u[8]; v16bf v; };

__device__ __forceinline__ __bf16 f2bf(float f) {
  unsigned u = __builtin_bit_cast(unsigned, f);
  unsigned r = (u + 0x7FFFu + ((u >> 16) & 1u)) >> 16;
  return __builtin_bit_cast(__bf16, (unsigned short)r);
}
__device__ __forceinline__ float bf2f(__bf16 h) {
  unsigned u = ((unsigned)__builtin_bit_cast(unsigned short, h)) << 16;
  return __builtin_bit_cast(float, u);
}
// hardware packed conversion: dst.lo = bf16(a), dst.hi = bf16(b)
__device__ __forceinline__ unsigned pack2bf(float a, float b) {
  unsigned r;
  asm("v_cvt_pk_bf16_f32 %0, %1, %2" : "=v"(r) : "v"(a), "v"(b));
  return r;
}
__device__ __forceinline__ float bflo(unsigned u) {
  return __builtin_bit_cast(float, u << 16);
}
__device__ __forceinline__ float bfhi(unsigned u) {
  return __builtin_bit_cast(float, u & 0xFFFF0000u);
}

// A-matrix (16x32 bf16) per-lane K index for element e (ISA 7.12.2).
__device__ __forceinline__ int amap(int e, int g) {
  int v = e >> 1, p = e & 1;
  return ((v < 4) ? (2 * v + 8 * g) : (16 + 2 * (v - 4) + 8 * g)) + p;
}
// inverse A-map for even kk within a 32-block: returns even element index
__device__ __forceinline__ int aemap(int kkl, int gg) {
  return (kkl < 16) ? (kkl - 8 * gg) : (kkl - 8 - 8 * gg);
}

__device__ __forceinline__ v8f wmma_bf16(v16bf a, v16bf b, v8f c) {
  return __builtin_amdgcn_wmma_f32_16x16x32_bf16(false, a, false, b, (short)0, c,
                                                 false, false);
}
__device__ __forceinline__ v16bf ldfrag(const __bf16* p) {
  return *(const v16bf*)p;                 // 32B: 2x b128 loads
}

// ---------------------------------------------------------------------------
// Pack f32 weight W(K,N) into fragment-ready bf16 (optionally row-scaled by
// (scale[k]+1)).  P[((kb*(N/16)+nb)*32+lane)*16+e] = W[kb*32+(e<8?e:e+8)+8*(lane>>4)][..]
// ---------------------------------------------------------------------------
__global__ __launch_bounds__(256)
void pack_b_kernel(const float* __restrict__ W, __bf16* __restrict__ P,
                   int K, int N) {
  int p = blockIdx.x * 256 + threadIdx.x;
  if (p >= K * N) return;
  int e = p & 15, lane = (p >> 4) & 31, rest = p >> 9;
  int nbT = N >> 4;
  int nb = rest % nbT, kb = rest / nbT;
  int kk = kb * 32 + ((e < 8) ? e : e + 8) + 8 * (lane >> 4);
  int n  = nb * 16 + (lane & 15);
  P[p] = f2bf(W[(size_t)kk * N + n]);
}
__global__ __launch_bounds__(256)
void pack_b_scaled_kernel(const float* __restrict__ W,
                          const float* __restrict__ scale,
                          __bf16* __restrict__ P, int K, int N) {
  int p = blockIdx.x * 256 + threadIdx.x;
  if (p >= K * N) return;
  int e = p & 15, lane = (p >> 4) & 31, rest = p >> 9;
  int nbT = N >> 4;
  int nb = rest % nbT, kb = rest / nbT;
  int kk = kb * 32 + ((e < 8) ? e : e + 8) + 8 * (lane >> 4);
  int n  = nb * 16 + (lane & 15);
  P[p] = f2bf(W[(size_t)kk * N + n] * (scale[kk] + 1.0f));
}

// ---------------------------------------------------------------------------
// Generic GEMM:  C(M,N) = op( rowscale * (A(M,K) @ W(K,N) + bias) )
// 256 threads (8 waves), tile 64x64, K-step 32; B pre-packed in frag layout.
// A staged via coalesced float4 loads + packed dword frag stores.
// ---------------------------------------------------------------------------
__global__ __launch_bounds__(256)
void gemm_bf16_kernel(const float* __restrict__ A, const __bf16* __restrict__ PB,
                      const float* __restrict__ bias,
                      const float* __restrict__ row_scale,
                      float* __restrict__ C, int M, int N, int K, int relu) {
  __shared__ __align__(32) __bf16 AsF[4 * 32 * 16];
  const int tid  = threadIdx.x;
  const int lane = tid & 31;
  const int w    = tid >> 5;
  const int g    = lane >> 4;
  const int l15  = lane & 15;
  const int m0   = blockIdx.y * 64;
  const int n0   = blockIdx.x * 64;
  const int tm   = w >> 1;
  const int tn0  = (2 * w) & 3;
  const int tn1  = tn0 + 1;
  const int nbT  = N >> 4;
  const int nb0  = (n0 >> 4) + tn0;
  const int nb1  = (n0 >> 4) + tn1;
  const int arow = tid >> 2;
  const int ac0  = (tid & 3) * 8;
  const int atm  = arow >> 4, al = arow & 15;

  v8f acc0 = {}; v8f acc1 = {};
  for (int k0 = 0; k0 < K; k0 += 32) {
    int kb = k0 >> 5;
    {
      const float4* fp = (const float4*)(A + (size_t)(m0 + arow) * K + k0 + ac0);
      float4 f0 = fp[0], f1 = fp[1];
      float fv[8] = {f0.x, f0.y, f0.z, f0.w, f1.x, f1.y, f1.z, f1.w};
#pragma unroll
      for (int p2 = 0; p2 < 4; ++p2) {
        int kk = ac0 + 2 * p2;
        int gg = (kk >> 3) & 1;
        *(unsigned*)&AsF[((atm * 32) + al + 16 * gg) * 16 + aemap(kk, gg)] =
            pack2bf(fv[2 * p2], fv[2 * p2 + 1]);
      }
    }
    __syncthreads();
    v16bf a  = ldfrag(&AsF[(tm * 32 + lane) * 16]);
    v16bf b0 = ldfrag(&PB[((size_t)(kb * nbT + nb0) * 32 + lane) * 16]);
    v16bf b1 = ldfrag(&PB[((size_t)(kb * nbT + nb1) * 32 + lane) * 16]);
    acc0 = wmma_bf16(a, b0, acc0);
    acc1 = wmma_bf16(a, b1, acc1);
    __syncthreads();
  }
#pragma unroll
  for (int r = 0; r < 8; ++r) {
    int m  = m0 + tm * 16 + r + 8 * g;
    float rs = row_scale ? row_scale[m] : 1.0f;
    int na = n0 + tn0 * 16 + l15;
    int nb = n0 + tn1 * 16 + l15;
    float c0 = acc0[r] + (bias ? bias[na] : 0.0f);
    float c1 = acc1[r] + (bias ? bias[nb] : 0.0f);
    if (relu) { c0 = fmaxf(c0, 0.f); c1 = fmaxf(c1, 0.f); }
    C[(size_t)m * N + na] = c0 * rs;
    C[(size_t)m * N + nb] = c1 * rs;
  }
}

// ---------------------------------------------------------------------------
// Small y-dependent FiLM vectors and ce = ye_add @ e_out.w + e_out.b
// ---------------------------------------------------------------------------
__global__ __launch_bounds__(256)
void y_film_kernel(const float* __restrict__ y,
                   const float* Wyem, const float* byem,
                   const float* Wyea, const float* byea,
                   const float* Wyxm, const float* byxm,
                   const float* Wyxa, const float* byxa,
                   const float* Weo, const float* beo,
                   float* ye_mul, float* ye_add, float* yx_mul, float* yx_add,
                   float* ce) {
  __shared__ float yea_s[512];
  int t = threadIdx.x;
  for (int idx = t; idx < 512; idx += 256) {
    int b = idx >> 8, c = idx & 255;
    float am = byem[c], aa = byea[c], xm = byxm[c], xa = byxa[c];
    for (int j = 0; j < 64; ++j) {
      float yv = y[b * 64 + j];
      am += yv * Wyem[j * 256 + c];
      aa += yv * Wyea[j * 256 + c];
      xm += yv * Wyxm[j * 256 + c];
      xa += yv * Wyxa[j * 256 + c];
    }
    ye_mul[idx] = am; ye_add[idx] = aa; yx_mul[idx] = xm; yx_add[idx] = xa;
    yea_s[idx] = aa;
  }
  __syncthreads();
  for (int idx = t; idx < 128; idx += 256) {
    int b = idx >> 6, nn = idx & 63;
    float acc = beo[nn];
    for (int c = 0; c < 256; ++c) acc += yea_s[b * 256 + c] * Weo[c * 64 + nn];
    ce[idx] = acc;
  }
}

// ---------------------------------------------------------------------------
// Fused edge kernel: one workgroup per (b,q) row, 512 threads (16 waves).
//   Yt[c][k] transposed bf16 tile; attention reads rows with b64 loads.
// ---------------------------------------------------------------------------
#define SMEM_SZ 143872
__global__ __launch_bounds__(512)
void edge_fused_kernel(const float* __restrict__ E_g,
                       const float* __restrict__ mask_g,
                       const float* __restrict__ Qp, const float* __restrict__ Kp,
                       const float* __restrict__ Vp,
                       const __bf16* __restrict__ PWmul, const float* __restrict__ bmul,
                       const __bf16* __restrict__ PWadd, const float* __restrict__ badd,
                       const __bf16* __restrict__ PWoutS,   // per-batch (ye2+1)-scaled
                       const float* __restrict__ ce_g,
                       const __bf16* __restrict__ PW1, const float* __restrict__ b1,
                       const __bf16* __restrict__ PW2, const float* __restrict__ b2,
                       const float* __restrict__ wN1, const float* __restrict__ bN1,
                       const float* __restrict__ wN2, const float* __restrict__ bN2,
                       float* __restrict__ Eo, float* __restrict__ Wv,
                       float* __restrict__ e_part) {
  __shared__ __align__(32) char smem[SMEM_SZ];
  // region 1 (66560B): Yt bf16[128][260] -> Epre f32[256][65] -> HsF -> Sf
  __bf16* Yt  = (__bf16*)smem;                 // [c 0..127][k 0..255], stride 260
  auto Epre   = (float(*)[65])smem;
  __bf16* HsF = (__bf16*)smem;                 // [16 tm][4 ks][32 lane][16 e]
  auto Sf     = (float(*)[65])smem;
  auto Er  = (__bf16(*)[66])(smem + 66560);    // 33792 B
  float* red = (float*)(smem + 100352);        // 2048 floats
  float* Qs  = (float*)(smem + 108544);        // 128
  float* Msk = (float*)(smem + 109056);        // 256 node_mask row
  float* Mb  = (float*)(smem + 110080);        // 256 additive softmax bias
  __bf16* EaF = (__bf16*)(smem + 111104);      // [16 tm][2 ks][32 lane][16 e]

  const int tid = threadIdx.x;
  const int bq  = blockIdx.x;
  const int b   = bq >> 8;
  const int q   = bq & 255;
  const float* Erow = E_g + (size_t)bq * 256 * 64;
  const float maskq = mask_g[b * 256 + q];

  const int lane = tid & 31, wv = tid >> 5;
  const int g = lane >> 4, l15 = lane & 15;

  // ---- phase 0: mask row, E pool partials, frag-ready E tile ----
  if (tid < 256) {
    float m = mask_g[b * 256 + tid];
    Msk[tid] = m;
    Mb[tid]  = (m != 0.f) ? 0.f : -1e30f;
  }
  {
    int ch = tid & 63, seg = tid >> 6;   // 8 segs x 32 k
    float s = 0.f, ss = 0.f, mn = 3.4e38f, mx = -3.4e38f;
    for (int kk = seg * 32; kk < seg * 32 + 32; ++kk) {
      float v = Erow[kk * 64 + ch];
      s += v; ss += v * v; mn = fminf(mn, v); mx = fmaxf(mx, v);
    }
    int base = (seg * 64 + ch) * 4;
    red[base + 0] = s; red[base + 1] = ss; red[base + 2] = mn; red[base + 3] = mx;
  }
  for (int i = tid; i < 2048; i += 512) {        // E as WMMA A fragments
    int row = i >> 3, c0 = (i & 7) * 8;
    const float4* fp = (const float4*)(Erow + (size_t)row * 64 + c0);
    float4 f0 = fp[0], f1 = fp[1];
    float fv[8] = {f0.x, f0.y, f0.z, f0.w, f1.x, f1.y, f1.z, f1.w};
    int tmv = row >> 4, l = row & 15;
#pragma unroll
    for (int p2 = 0; p2 < 4; ++p2) {
      int kk = c0 + 2 * p2;
      int ks = kk >> 5, kkl = kk & 31;
      int gg = (kkl >> 3) & 1;
      *(unsigned*)&EaF[((tmv * 2 + ks) * 32 + l + 16 * gg) * 16 + aemap(kkl, gg)] =
          pack2bf(fv[2 * p2], fv[2 * p2 + 1]);
    }
  }
  __syncthreads();
  if (tid < 256) {
    int ch = tid & 63, st = tid >> 6;
    float acc = (st == 2) ? 3.4e38f : ((st == 3) ? -3.4e38f : 0.f);
    for (int sgi = 0; sgi < 8; ++sgi) {
      float v = red[(sgi * 64 + ch) * 4 + st];
      if (st == 2) acc = fminf(acc, v);
      else if (st == 3) acc = fmaxf(acc, v);
      else acc += v;
    }
    e_part[(size_t)(bq * 4 + st) * 64 + ch] = acc;
  }
  __syncthreads();

  v8f eacc[4] = {};   // newE accumulators, persistent across chunks

  for (int cc = 0; cc < 2; ++cc) {
    const int cbase = cc * 128;
    if (tid < 128) Qs[tid] = Qp[(size_t)bq * 256 + cbase + tid];
    __syncthreads();

    // --- E1/E2 GEMM + Y build (transposed store) ---
    {
      v16bf a0 = ldfrag(&EaF[((wv * 2 + 0) * 32 + lane) * 16]);
      v16bf a1 = ldfrag(&EaF[((wv * 2 + 1) * 32 + lane) * 16]);
#pragma unroll 2
      for (int tn = 0; tn < 8; ++tn) {
        int nb = cc * 8 + tn;                 // nbT = 16
        int nc = cbase + tn * 16 + l15;
        v16bf bm0 = ldfrag(&PWmul[((size_t)(0 * 16 + nb) * 32 + lane) * 16]);
        v16bf bm1 = ldfrag(&PWmul[((size_t)(1 * 16 + nb) * 32 + lane) * 16]);
        v16bf ba0 = ldfrag(&PWadd[((size_t)(0 * 16 + nb) * 32 + lane) * 16]);
        v16bf ba1 = ldfrag(&PWadd[((size_t)(1 * 16 + nb) * 32 + lane) * 16]);
        v8f acc1 = {}; v8f acc2 = {};
        acc1 = wmma_bf16(a0, bm0, acc1);
        acc1 = wmma_bf16(a1, bm1, acc1);
        acc2 = wmma_bf16(a0, ba0, acc2);
        acc2 = wmma_bf16(a1, ba1, acc2);
        float qv = Qs[tn * 16 + l15];
        float bm = bmul[nc], ba = badd[nc];
        __bf16* yrow = Yt + (size_t)(tn * 16 + l15) * 260;
#pragma unroll
        for (int rp = 0; rp < 4; ++rp) {
          int kr0 = wv * 16 + 8 * g + 2 * rp;
          float em0 = maskq * Msk[kr0], em1 = maskq * Msk[kr0 + 1];
          float e1a = (acc1[2 * rp] + bm) * em0;
          float e1b = (acc1[2 * rp + 1] + bm) * em1;
          float e2a = (acc2[2 * rp] + ba) * em0;
          float e2b = (acc2[2 * rp + 1] + ba) * em1;
          float qk0 = qv * Kp[((size_t)(b * 256 + kr0)) * 256 + nc] * 0.17677669529663687f;
          float qk1 = qv * Kp[((size_t)(b * 256 + kr0 + 1)) * 256 + nc] * 0.17677669529663687f;
          *(unsigned*)(yrow + kr0) =
              pack2bf(qk0 * (e1a + 1.0f) + e2a, qk1 * (e1b + 1.0f) + e2b);
        }
      }
    }
    __syncthreads();

    // --- attention: softmax over k per channel (vector b64 row reads) ---
    {
      int c = tid & 127, seg = tid >> 7;  // 4 segs x 64 k
      int chg = cbase + c;
      const uint2* yrowp = (const uint2*)(Yt + (size_t)c * 260 + seg * 64);
      float mp = -3.4e38f;
#pragma unroll
      for (int j = 0; j < 16; ++j) {
        uint2 u = yrowp[j];
        int kb = seg * 64 + j * 4;
        float v0 = bflo(u.x) + Mb[kb],     v1 = bfhi(u.x) + Mb[kb + 1];
        float v2 = bflo(u.y) + Mb[kb + 2], v3 = bfhi(u.y) + Mb[kb + 3];
        mp = fmaxf(mp, fmaxf(fmaxf(v0, v1), fmaxf(v2, v3)));
      }
      red[seg * 128 + c] = mp;
      __syncthreads();
      float mx = fmaxf(fmaxf(red[c], red[128 + c]), fmaxf(red[256 + c], red[384 + c]));
      float sp = 0.f, wp = 0.f;
#pragma unroll
      for (int j = 0; j < 16; ++j) {
        uint2 u = yrowp[j];
        int kb = seg * 64 + j * 4;
        float t0 = __expf(bflo(u.x) + Mb[kb]     - mx);
        float t1 = __expf(bfhi(u.x) + Mb[kb + 1] - mx);
        float t2 = __expf(bflo(u.y) + Mb[kb + 2] - mx);
        float t3 = __expf(bfhi(u.y) + Mb[kb + 3] - mx);
        sp += t0 + t1 + t2 + t3;
        wp += t0 * Vp[((size_t)(b * 256 + kb)) * 256 + chg]
            + t1 * Vp[((size_t)(b * 256 + kb + 1)) * 256 + chg]
            + t2 * Vp[((size_t)(b * 256 + kb + 2)) * 256 + chg]
            + t3 * Vp[((size_t)(b * 256 + kb + 3)) * 256 + chg];
      }
      red[512 + seg * 128 + c] = sp;
      red[1024 + seg * 128 + c] = wp;
      __syncthreads();
      if (seg == 0) {
        float sd = red[512 + c] + red[640 + c] + red[768 + c] + red[896 + c];
        float wn = red[1024 + c] + red[1152 + c] + red[1280 + c] + red[1408 + c];
        Wv[(size_t)bq * 256 + chg] = wn / sd;
      }
    }
    __syncthreads();

    // --- newE partial: Y @ (scaled e_out.w), K=128 this chunk ---
    {
      int m = wv * 16 + l15;
      const __bf16* PWb = PWoutS + (size_t)b * 16384;
#pragma unroll
      for (int ks = 0; ks < 4; ++ks) {
        v16bf a;
#pragma unroll
        for (int e = 0; e < 16; ++e) {
          int kk = ks * 32 + amap(e, g);           // chunk-local channel
          a[e] = Yt[(size_t)kk * 260 + m];
        }
#pragma unroll
        for (int tn = 0; tn < 4; ++tn) {           // nbT = 4
          v16bf bb = ldfrag(&PWb[((size_t)((cc * 4 + ks) * 4 + tn) * 32 + lane) * 16]);
          eacc[tn] = wmma_bf16(a, bb, eacc[tn]);
        }
      }
    }
    __syncthreads();
  }

  // ---- phase 2a: Epre = E + newE*e_mask ----
  {
#pragma unroll
    for (int tn = 0; tn < 4; ++tn) {
      int nn = tn * 16 + l15;
      float cev = ce_g[b * 64 + nn];
#pragma unroll
      for (int r = 0; r < 8; ++r) {
        int kr = wv * 16 + r + 8 * g;
        float em = maskq * Msk[kr];
        Epre[kr][nn] = Erow[kr * 64 + nn] + (eacc[tn][r] + cev) * em;
      }
    }
  }
  __syncthreads();

  // ---- phase 2b: LN1 -> Er (bf16, packed dword writes) ----
  if (tid < 256) {
    int kr = tid;
    float s = 0.f, ss = 0.f;
    for (int ch = 0; ch < 64; ++ch) { float v = Epre[kr][ch]; s += v; ss += v * v; }
    float mu = s * (1.f / 64.f);
    float rstd = rsqrtf(ss * (1.f / 64.f) - mu * mu + 1e-5f);
    for (int ch = 0; ch < 64; ch += 2) {
      float v0 = (Epre[kr][ch]     - mu) * rstd * wN1[ch]     + bN1[ch];
      float v1 = (Epre[kr][ch + 1] - mu) * rstd * wN1[ch + 1] + bN1[ch + 1];
      *(unsigned*)&Er[kr][ch] = pack2bf(v0, v1);
    }
  }
  __syncthreads();

  // ---- phase 2c: H = relu(Er @ linE1 + b1); store H in A-frag layout ----
  {
    int m = wv * 16 + l15;
    FragU ua0, ua1;
#pragma unroll
    for (int ep = 0; ep < 8; ++ep) {
      int kkl = amap(2 * ep, g);                   // even
      ua0.u[ep] = *(const unsigned*)&Er[m][kkl];
      ua1.u[ep] = *(const unsigned*)&Er[m][32 + kkl];
    }
    v16bf a0 = ua0.v, a1 = ua1.v;
    for (int tn = 0; tn < 8; ++tn) {               // PW1 nbT = 8
      v16bf bb0 = ldfrag(&PW1[((size_t)(0 * 8 + tn) * 32 + lane) * 16]);
      v16bf bb1 = ldfrag(&PW1[((size_t)(1 * 8 + tn) * 32 + lane) * 16]);
      v8f acc = {};
      acc = wmma_bf16(a0, bb0, acc);
      acc = wmma_bf16(a1, bb1, acc);
      float bv = b1[tn * 16 + l15];
      int n = tn * 16 + l15;
      int kk = n & 31, ks2 = n >> 5;
      int g2 = (kk >> 3) & 1;
      int e2 = (kk < 16) ? (kk - 8 * g2) : (kk - 8 - 8 * g2);
#pragma unroll
      for (int r = 0; r < 8; ++r) {
        int lane2 = (r + 8 * g) + 16 * g2;
        HsF[((wv * 4 + ks2) * 32 + lane2) * 16 + e2] = f2bf(fmaxf(acc[r] + bv, 0.f));
      }
    }
  }
  __syncthreads();

  // ---- phase 2d: F = H @ linE2 + b2;  S = Er + F ----
  {
    v8f facc[4] = {};
#pragma unroll
    for (int ks = 0; ks < 4; ++ks) {
      v16bf a = ldfrag(&HsF[((wv * 4 + ks) * 32 + lane) * 16]);
#pragma unroll
      for (int tn = 0; tn < 4; ++tn) {             // PW2 nbT = 4
        v16bf bb = ldfrag(&PW2[((size_t)(ks * 4 + tn) * 32 + lane) * 16]);
        facc[tn] = wmma_bf16(a, bb, facc[tn]);
      }
    }
    __syncthreads();   // H reads done before S overwrites region 1
#pragma unroll
    for (int tn = 0; tn < 4; ++tn) {
      int nn = tn * 16 + l15;
#pragma unroll
      for (int r = 0; r < 8; ++r) {
        int kr = wv * 16 + r + 8 * g;
        Sf[kr][nn] = bf2f(Er[kr][nn]) + facc[tn][r] + b2[nn];
      }
    }
  }
  __syncthreads();

  // ---- phase 2e: LN2 -> Eo ----
  if (tid < 256) {
    int kr = tid;
    float s = 0.f, ss = 0.f;
    for (int ch = 0; ch < 64; ++ch) { float v = Sf[kr][ch]; s += v; ss += v * v; }
    float mu = s * (1.f / 64.f);
    float rstd = rsqrtf(ss * (1.f / 64.f) - mu * mu + 1e-5f);
    size_t base = ((size_t)bq * 256 + kr) * 64;
    for (int ch = 0; ch < 64; ++ch)
      Eo[base + ch] = (Sf[kr][ch] - mu) * rstd * wN2[ch] + bN2[ch];
  }
}

// ---------------------------------------------------------------------------
// Global y update: pools + MLPs + LN + FFN. One block, 256 threads.
// ---------------------------------------------------------------------------
__global__ __launch_bounds__(256)
void y_update_kernel(const float* __restrict__ X, const float* __restrict__ y,
                     const float* __restrict__ e_part,
                     const float* Wyy, const float* byy,
                     const float* Wxy, const float* bxy,
                     const float* Wey, const float* bey,
                     const float* Wo1, const float* bo1,
                     const float* Wo2, const float* bo2,
                     const float* wn1, const float* bn1,
                     const float* Wl1, const float* bl1,
                     const float* Wl2, const float* bl2,
                     const float* wn2, const float* bn2,
                     float* __restrict__ yo) {
  __shared__ float xpool[2][1024];
  __shared__ float epool[2][256];
  __shared__ float ny[2][64];
  __shared__ float tmp[2][64];
  __shared__ float yr[2][64];
  __shared__ float mv[2][2];
  __shared__ float hb[2][2048];
  int t = threadIdx.x;
  for (int b = 0; b < 2; ++b) {
    float s = 0.f, ss = 0.f, mn = 3.4e38f, mx = -3.4e38f;
    for (int nn = 0; nn < 256; ++nn) {
      float v = X[((size_t)b * 256 + nn) * 256 + t];
      s += v; ss += v * v; mn = fminf(mn, v); mx = fmaxf(mx, v);
    }
    float mu = s * (1.f / 256.f);
    float sd = sqrtf(fmaxf(0.f, ss - 256.f * mu * mu) / 255.f);
    xpool[b][t] = mu; xpool[b][256 + t] = mn; xpool[b][512 + t] = mx; xpool[b][768 + t] = sd;
  }
  if (t < 128) {
    int b = t >> 6, ch = t & 63;
    float s = 0.f, ss = 0.f, mn = 3.4e38f, mx = -3.4e38f;
    for (int qq = 0; qq < 256; ++qq) {
      size_t bq4 = (size_t)(b * 256 + qq) * 4;
      s  += e_part[(bq4 + 0) * 64 + ch];
      ss += e_part[(bq4 + 1) * 64 + ch];
      mn = fminf(mn, e_part[(bq4 + 2) * 64 + ch]);
      mx = fmaxf(mx, e_part[(bq4 + 3) * 64 + ch]);
    }
    const float N = 65536.f;
    float mu = s / N;
    float sd = sqrtf(fmaxf(0.f, ss - N * mu * mu) / (N - 1.f));
    epool[b][ch] = mu; epool[b][64 + ch] = mn; epool[b][128 + ch] = mx; epool[b][192 + ch] = sd;
  }
  __syncthreads();
  if (t < 128) {
    int b = t >> 6, d = t & 63;
    float acc = byy[d] + bxy[d] + bey[d];
    for (int j = 0; j < 64; ++j)   acc += y[b * 64 + j] * Wyy[j * 64 + d];
    for (int j = 0; j < 1024; ++j) acc += xpool[b][j] * Wxy[j * 64 + d];
    for (int j = 0; j < 256; ++j)  acc += epool[b][j] * Wey[j * 64 + d];
    ny[b][d] = acc;
  }
  __syncthreads();
  if (t < 128) {
    int b = t >> 6, d = t & 63;
    float acc = bo1[d];
    for (int j = 0; j < 64; ++j) acc += ny[b][j] * Wo1[j * 64 + d];
    tmp[b][d] = fmaxf(acc, 0.f);
  }
  __syncthreads();
  if (t < 128) {
    int b = t >> 6, d = t & 63;
    float acc = bo2[d];
    for (int j = 0; j < 64; ++j) acc += tmp[b][j] * Wo2[j * 64 + d];
    ny[b][d] = y[b * 64 + d] + acc;
  }
  __syncthreads();
  if (t < 2) {
    float s = 0.f, ss = 0.f;
    for (int j = 0; j < 64; ++j) { float v = ny[t][j]; s += v; ss += v * v; }
    float mu = s * (1.f / 64.f);
    mv[t][0] = mu; mv[t][1] = rsqrtf(ss * (1.f / 64.f) - mu * mu + 1e-5f);
  }
  __syncthreads();
  if (t < 128) {
    int b = t >> 6, d = t & 63;
    yr[b][d] = (ny[b][d] - mv[b][0]) * mv[b][1] * wn1[d] + bn1[d];
  }
  __syncthreads();
  for (int idx = t; idx < 4096; idx += 256) {
    int b = idx >> 11, f = idx & 2047;
    float acc = bl1[f];
    for (int j = 0; j < 64; ++j) acc += yr[b][j] * Wl1[j * 2048 + f];
    hb[b][f] = fmaxf(acc, 0.f);
  }
  __syncthreads();
  if (t < 128) {
    int b = t >> 6, d = t & 63;
    float acc = bl2[d];
    for (int j = 0; j < 2048; ++j) acc += hb[b][j] * Wl2[j * 64 + d];
    ny[b][d] = yr[b][d] + acc;
  }
  __syncthreads();
  if (t < 2) {
    float s = 0.f, ss = 0.f;
    for (int j = 0; j < 64; ++j) { float v = ny[t][j]; s += v; ss += v * v; }
    float mu = s * (1.f / 64.f);
    mv[t][0] = mu; mv[t][1] = rsqrtf(ss * (1.f / 64.f) - mu * mu + 1e-5f);
  }
  __syncthreads();
  if (t < 128) {
    int b = t >> 6, d = t & 63;
    yo[t] = (ny[b][d] - mv[b][0]) * mv[b][1] * wn2[d] + bn2[d];
  }
}

// ---------------------------------------------------------------------------
__global__ __launch_bounds__(256)
void film_x_kernel(const float* __restrict__ Wv, const float* __restrict__ yx_add,
                   const float* __restrict__ yx_mul, float* __restrict__ A1,
                   int total) {
  int idx = blockIdx.x * 256 + threadIdx.x;
  if (idx >= total) return;
  int row = idx >> 8;
  int b = row >> 8;
  int c = idx & 255;
  A1[idx] = yx_add[b * 256 + c] + (yx_mul[b * 256 + c] + 1.f) * Wv[idx];
}

// ---------------------------------------------------------------------------
__global__ __launch_bounds__(256)
void ln_res_kernel(const float* __restrict__ x, const float* __restrict__ addv,
                   const float* __restrict__ w, const float* __restrict__ b,
                   float* __restrict__ out) {
  __shared__ float red[256];
  const int row = blockIdx.x;
  const int t = threadIdx.x;
  float val = x[(size_t)row * 256 + t] + addv[(size_t)row * 256 + t];
  red[t] = val;
  __syncthreads();
  for (int s = 128; s > 0; s >>= 1) { if (t < s) red[t] += red[t + s]; __syncthreads(); }
  float mu = red[0] * (1.f / 256.f);
  __syncthreads();
  red[t] = (val - mu) * (val - mu);
  __syncthreads();
  for (int s = 128; s > 0; s >>= 1) { if (t < s) red[t] += red[t + s]; __syncthreads(); }
  float var = red[0] * (1.f / 256.f);
  out[(size_t)row * 256 + t] = (val - mu) * rsqrtf(var + 1e-5f) * w[t] + b[t];
}

// ---------------------------------------------------------------------------
extern "C" void kernel_launch(void* const* d_in, const int* in_sizes, int n_in,
                              void* d_out, int out_size, void* d_ws, size_t ws_size,
                              hipStream_t stream) {
  (void)in_sizes; (void)n_in; (void)out_size; (void)ws_size;
  const float* X    = (const float*)d_in[0];
  const float* E    = (const float*)d_in[1];
  const float* y    = (const float*)d_in[2];
  const float* mask = (const float*)d_in[3];
  auto P = [&](int i) { return (const float*)d_in[i]; };
  const float *Wq = P(4),  *bq = P(5),  *Wk = P(6),  *bk = P(7);
  const float *Wvp = P(8), *bvp = P(9);
  const float *Wea = P(10), *bea = P(11), *Wem = P(12), *bem = P(13);
  const float *Wyem = P(14), *byem = P(15), *Wyea = P(16), *byea = P(17);
  const float *Wyxm = P(18), *byxm = P(19), *Wyxa = P(20), *byxa = P(21);
  const float *Wyy = P(22), *byy = P(23), *Wxy = P(24), *bxy = P(25);
  const float *Wey = P(26), *bey = P(27);
  const float *Wxo = P(28), *bxo = P(29), *Weo = P(30), *beo = P(31);
  const float *Wo1 = P(32), *bo1 = P(33), *Wo2 = P(34), *bo2 = P(35);
  const float *WX1 = P(36), *bX1 = P(37), *WX2 = P(38), *bX2 = P(39);
  const float *WE1 = P(40), *bE1 = P(41), *WE2 = P(42), *bE2 = P(43);
  const float *Wy1 = P(44), *by1 = P(45), *Wy2 = P(46), *by2 = P(47);
  const float *wNX1 = P(48), *bNX1 = P(49), *wNX2 = P(50), *bNX2 = P(51);
  const float *wNE1 = P(52), *bNE1 = P(53), *wNE2 = P(54), *bNE2 = P(55);
  const float *wNy1 = P(56), *bNy1 = P(57), *wNy2 = P(58), *bNy2 = P(59);

  float* out = (float*)d_out;
  float* Xo = out;
  float* Eo = out + 131072;
  float* yo = out + 131072 + 8388608;

  float* ws = (float*)d_ws;
  float* Qp     = ws;               // 131072
  float* Kp     = ws + 131072;
  float* Vp     = ws + 262144;
  float* Wvbuf  = ws + 393216;
  float* e_part = ws + 524288;      // 512*4*64
  float* ye_mul = ws + 655360;
  float* ye_add = ws + 655872;
  float* yx_mul = ws + 656384;
  float* yx_add = ws + 656896;
  float* ce     = ws + 657408;
  float* A1     = ws + 657536;
  float* newX   = ws + 788608;
  float* Xr     = ws + 919680;
  float* Hx     = ws + 1050752;     // 512*2048
  float* Fx     = ws + 2099328;
  __bf16* PWq   = (__bf16*)(ws + 2230400);  // 65536 bf16
  __bf16* PWk   = (__bf16*)(ws + 2263168);
  __bf16* PWv   = (__bf16*)(ws + 2295936);
  __bf16* PWxo  = (__bf16*)(ws + 2328704);
  __bf16* PWX1  = (__bf16*)(ws + 2361472);  // 524288 bf16
  __bf16* PWX2  = (__bf16*)(ws + 2623616);
  __bf16* PWmul = (__bf16*)(ws + 2885760);  // 16384 bf16
  __bf16* PWadd = (__bf16*)(ws + 2893952);
  __bf16* PW1   = (__bf16*)(ws + 2902144);  // 8192 bf16
  __bf16* PW2   = (__bf16*)(ws + 2906240);
  __bf16* PWoutS= (__bf16*)(ws + 2910336);  // 2 x 16384 bf16 (per-batch scaled)

  auto packB = [&](const float* W, __bf16* Pb, int K, int N) {
    pack_b_kernel<<<(K * N + 255) / 256, 256, 0, stream>>>(W, Pb, K, N);
  };
  packB(Wq, PWq, 256, 256);   packB(Wk, PWk, 256, 256);
  packB(Wvp, PWv, 256, 256);  packB(Wxo, PWxo, 256, 256);
  packB(WX1, PWX1, 256, 2048); packB(WX2, PWX2, 2048, 256);
  packB(Wem, PWmul, 64, 256); packB(Wea, PWadd, 64, 256);
  packB(WE1, PW1, 64, 128);   packB(WE2, PW2, 128, 64);

  // small y-FiLM vectors, then per-batch (ye_mul+1)-scaled e_out pack
  y_film_kernel<<<1, 256, 0, stream>>>(y, Wyem, byem, Wyea, byea, Wyxm, byxm,
                                       Wyxa, byxa, Weo, beo,
                                       ye_mul, ye_add, yx_mul, yx_add, ce);
  pack_b_scaled_kernel<<<64, 256, 0, stream>>>(Weo, ye_mul,       PWoutS,         256, 64);
  pack_b_scaled_kernel<<<64, 256, 0, stream>>>(Weo, ye_mul + 256, PWoutS + 16384, 256, 64);

  gemm_bf16_kernel<<<dim3(4, 8), 256, 0, stream>>>(X, PWq, bq,  mask, Qp, 512, 256, 256, 0);
  gemm_bf16_kernel<<<dim3(4, 8), 256, 0, stream>>>(X, PWk, bk,  mask, Kp, 512, 256, 256, 0);
  gemm_bf16_kernel<<<dim3(4, 8), 256, 0, stream>>>(X, PWv, bvp, mask, Vp, 512, 256, 256, 0);

  edge_fused_kernel<<<512, 512, 0, stream>>>(E, mask, Qp, Kp, Vp,
                                             PWmul, bem, PWadd, bea, PWoutS, ce,
                                             PW1, bE1, PW2, bE2,
                                             wNE1, bNE1, wNE2, bNE2,
                                             Eo, Wvbuf, e_part);

  y_update_kernel<<<1, 256, 0, stream>>>(X, y, e_part, Wyy, byy, Wxy, bxy,
                                         Wey, bey, Wo1, bo1, Wo2, bo2,
                                         wNy1, bNy1, Wy1, by1, Wy2, by2,
                                         wNy2, bNy2, yo);

  film_x_kernel<<<512, 256, 0, stream>>>(Wvbuf, yx_add, yx_mul, A1, 131072);
  gemm_bf16_kernel<<<dim3(4, 8), 256, 0, stream>>>(A1, PWxo, bxo, mask, newX, 512, 256, 256, 0);
  ln_res_kernel<<<512, 256, 0, stream>>>(X, newX, wNX1, bNX1, Xr);
  gemm_bf16_kernel<<<dim3(32, 8), 256, 0, stream>>>(Xr, PWX1, bX1, nullptr, Hx, 512, 2048, 256, 1);
  gemm_bf16_kernel<<<dim3(4, 8), 256, 0, stream>>>(Hx, PWX2, bX2, nullptr, Fx, 512, 256, 2048, 0);
  ln_res_kernel<<<512, 256, 0, stream>>>(Xr, Fx, wNX2, bNX2, Xo);
}
